// SparseGCNBlock_47863115547046
// MI455X (gfx1250) — matrix operand order repaired and checked by reading, not compiled, not run on hardware
//
#include <hip/hip_runtime.h>

// Fused SparseGCNBlock for MI455X (gfx1250, wave32, WMMA + TDM):
//   x tile staged by TENSOR_LOAD_TO_LDS (TDM DMA, LDS pad = bank-padded rows)
//   h = x@W   (f16 WMMA, f32 accumulate, tiles in LDS)
//   deg/rsqrt-normalized scatter-add over graph-local edges (LDS f32 atomics)
//   + bias + residual (from LDS), LayerNorm(C=128), ReLU -> out
// One workgroup = 2 graphs = 56 rows (padded to 64 = 4 WMMA M-tiles).

#define C_DIM 128
#define NPG   28
#define EPG   56
#define GPB   2                 // graphs per block
#define MROWS (GPB * NPG)       // 56 real rows
#define MPAD  64                // padded to 4 x 16 M-tiles
#define EB    (GPB * EPG)       // 112 edges per block
#define WTS   136               // f16 halves per row (W^T), bank-padded
#define XSF   132               // f32 per row (staged x), bank-padded (TDM pad: 4 dw / 128 dw)
#define HS    132               // f32 per row (h / agg), bank-padded
#define LN_EPS 1e-5f

typedef __attribute__((ext_vector_type(16))) _Float16     v16h;
typedef __attribute__((ext_vector_type(8)))  _Float16     v8h;
typedef __attribute__((ext_vector_type(8)))  float        v8f;
typedef __attribute__((ext_vector_type(4)))  unsigned int u32x4;
typedef __attribute__((ext_vector_type(8)))  int          i32x8;
typedef __attribute__((ext_vector_type(4)))  int          i32x4;

__global__ __launch_bounds__(256, 1)
void gcn_block_fused_wmma(const float* __restrict__ x,
                          const int*   __restrict__ ei,
                          const float* __restrict__ Wm,
                          const float* __restrict__ bias,
                          const float* __restrict__ ew,
                          const float* __restrict__ gamma_,
                          const float* __restrict__ beta_,
                          float*       __restrict__ out,
                          int Etot)
{
    __shared__ _Float16 lWt[C_DIM * WTS];   // W transposed, f16: 34816 B
    __shared__ float    lXf [MPAD * XSF];   // x tile, f32:       33792 B (TDM dest)
    __shared__ float    lH  [MPAD * HS];    // h = x@W, f32:      33792 B
    __shared__ float    lAgg[MPAD * HS];    // aggregated, f32:   33792 B
    __shared__ float    lDeg [MROWS];
    __shared__ float    lDinv[MROWS];
    __shared__ float    lEw  [EB];
    __shared__ unsigned short lEdge[EB];    // (dst<<8)|src, graph-local

    const int tid  = threadIdx.x;
    const int lane = tid & 31;
    const int wv   = tid >> 5;              // 8 waves
    const int g0i  = blockIdx.x * GPB;
    const int row0 = g0i * NPG;
    const long long e0 = (long long)g0i * EPG;

#if __has_builtin(__builtin_amdgcn_tensor_load_to_lds)
    // ---- TDM: DMA the 56x128 f32 x tile into LDS with 4-DWORD row padding ----
    // D# per CDNA5 ISA: group0 = {flags, lds_addr, gaddr}, type=2
    // group1: data_size=4B, pad_enable, pad_interval=128 dw (code 6),
    //         pad_amount=4 dw (code 3) -> LDS row stride 132 floats.
    if (tid == 0) {
        unsigned long long ga = (unsigned long long)(uintptr_t)(x + (long long)row0 * C_DIM);
        unsigned lds_off = (unsigned)(uintptr_t)(&lXf[0]);  // flat LDS addr low 32b = LDS offset
        u32x4 g0;
        g0[0] = 1u;                                           // count=1 (valid), user mode
        g0[1] = lds_off;                                      // lds_addr
        g0[2] = (unsigned)(ga & 0xFFFFFFFFu);                 // global_addr[31:0]
        g0[3] = (unsigned)((ga >> 32) & 0x1FFFFFFu) | (2u << 30); // addr[56:32] | type=2
        i32x8 g1;
        g1[0] = (2 << 16) | (1 << 20) | (6 << 22) | (3 << 25); // data_size=4B|pad_en|intv|amt
        g1[1] = (C_DIM << 16);     // abar_addr=0 | tensor_dim0[15:0] = 128
        g1[2] = (MROWS << 16);     // tensor_dim0[31:16]=0 | tensor_dim1[15:0] = 56
        g1[3] = (C_DIM << 16);     // tensor_dim1[31:16]=0 | tile_dim0 = 128
        g1[4] = MROWS;             // tile_dim1 = 56 | tile_dim2 = 0
        g1[5] = C_DIM;             // tensor_dim0_stride[31:0] = 128 elements
        g1[6] = 0;                 // dim0_stride[47:32]=0 | dim1_stride[15:0]=0
        g1[7] = 0;
        i32x4 gz4 = {0, 0, 0, 0};
        i32x8 gz8 = {0, 0, 0, 0, 0, 0, 0, 0};
        // 6-arg form (clang-23 / therock-10.0 headers): groups 0..3 + extra + cpol
        __builtin_amdgcn_tensor_load_to_lds(g0, g1, gz4, gz4, gz8, 0);
    }
#else
    // ---- fallback: manual coalesced staging of x tile (f32) ----
    {
        const float* xb = x + (long long)row0 * C_DIM;
        for (int i = tid; i < MROWS * C_DIM; i += 256) {
            int r = i >> 7, c = i & (C_DIM - 1);
            lXf[r * XSF + c] = xb[i];
        }
    }
#endif

    // ---- stage W^T as f16 into LDS (coalesced global reads) ----
    for (int i = tid; i < C_DIM * C_DIM; i += 256) {
        int k = i >> 7, n = i & (C_DIM - 1);
        lWt[n * WTS + k] = (_Float16)Wm[i];
    }
    // ---- zero the 8 pad rows of the x tile (rows 56..63) ----
    for (int i = tid; i < (MPAD - MROWS) * XSF; i += 256)
        lXf[MROWS * XSF + i] = 0.f;

    // ---- stage graph-local edges + degree init (self loop weight 1) ----
    int   myD = 0; float myW = 0.f; const int haveE = (tid < EB);
    if (haveE) {
        long long ge = e0 + tid;
        int s = ei[ge] - row0;
        int d = ei[(long long)Etot + ge] - row0;
        myD = d;
        myW = ew[tid % EPG];
        lEdge[tid] = (unsigned short)((d << 8) | (s & 255));
        lEw[tid]   = myW;
    }
    if (tid < MROWS) lDeg[tid] = 1.0f;

#if __has_builtin(__builtin_amdgcn_tensor_load_to_lds)
    if (tid == 0) __builtin_amdgcn_s_wait_tensorcnt(0);  // x tile resident in LDS
#endif
    __syncthreads();

    if (haveE) atomicAdd(&lDeg[myD], myW);
    __syncthreads();

    if (tid < MROWS) lDinv[tid] = rsqrtf(lDeg[tid]);  // deg >= 1 always

    // ---- GEMM: lH[0:64,0:128] = lXf @ W  via v_wmma_f32_16x16x32_f16 ----
    {
        const int ntb  = wv << 4;             // this wave's 16-wide N column
        const int rsel = lane & 15;
        const int hiA  = (lane >> 4) << 3;    // A: +8 K for lanes 16..31
        const int hiB  = (lane >> 4) << 3;    // B: +8 K for lanes 16..31
        v8f acc[4] = {};

        #pragma unroll
        for (int kt = 0; kt < 4; ++kt) {
            // B fragment: lane holds column ntb+rsel, K=kb+0..7 and kb+16..23 (same as A)
            const _Float16* pw = &lWt[(ntb + rsel) * WTS + (kt << 5) + hiB];
            v8h b0 = *(const v8h*)pw;
            v8h b1 = *(const v8h*)(pw + 16);
            v16h bf;
            #pragma unroll
            for (int j = 0; j < 8; ++j) { bf[j] = b0[j]; bf[8 + j] = b1[j]; }

            #pragma unroll
            for (int mt = 0; mt < 4; ++mt) {
                // A fragment (f32 -> f16): row mt*16+rsel;
                // halves 0-7 -> K=kb+0..7, halves 8-15 -> K=kb+16..23 (kb = kt*32 + hiA)
                const float* pa = &lXf[((mt << 4) + rsel) * XSF + (kt << 5) + hiA];
                float4 a0 = *(const float4*)(pa +  0);
                float4 a1 = *(const float4*)(pa +  4);
                float4 a2 = *(const float4*)(pa + 16);
                float4 a3 = *(const float4*)(pa + 20);
                v16h af;
                af[0]  = (_Float16)a0.x; af[1]  = (_Float16)a0.y;
                af[2]  = (_Float16)a0.z; af[3]  = (_Float16)a0.w;
                af[4]  = (_Float16)a1.x; af[5]  = (_Float16)a1.y;
                af[6]  = (_Float16)a1.z; af[7]  = (_Float16)a1.w;
                af[8]  = (_Float16)a2.x; af[9]  = (_Float16)a2.y;
                af[10] = (_Float16)a2.z; af[11] = (_Float16)a2.w;
                af[12] = (_Float16)a3.x; af[13] = (_Float16)a3.y;
                af[14] = (_Float16)a3.z; af[15] = (_Float16)a3.w;

                acc[mt] = __builtin_amdgcn_wmma_f32_16x16x32_f16(
                    false, af, false, bf, (short)0, acc[mt], false, false);
            }
        }
        // D writeback: VGPR j -> M = j (lanes 0-15) / j+8 (lanes 16-31)
        #pragma unroll
        for (int mt = 0; mt < 4; ++mt) {
            #pragma unroll
            for (int j = 0; j < 8; ++j) {
                int r = (mt << 4) + j + hiA;
                lH[r * HS + ntb + rsel] = acc[mt][j];
            }
        }
    }
    __syncthreads();

    // ---- agg init with self-loop term h[i] * dinv[i]^2 ----
    for (int i = tid; i < MROWS * C_DIM; i += 256) {
        int r = i >> 7, c = i & (C_DIM - 1);
        float di = lDinv[r];
        lAgg[r * HS + c] = lH[r * HS + c] * (di * di);
    }
    __syncthreads();

    // ---- edge scatter: agg[dst] += norm * h[src] (LDS f32 atomics) ----
    for (int t = tid; t < EB * 32; t += 256) {
        int e  = t >> 5;
        int cg = (t & 31) << 2;
        unsigned p = lEdge[e];
        int s = p & 255, d = p >> 8;
        float nrm = lDinv[s] * lEw[e] * lDinv[d];
        float4 hv = *(const float4*)&lH[s * HS + cg];
        float* ap = &lAgg[d * HS + cg];
        atomicAdd(ap + 0, hv.x * nrm);
        atomicAdd(ap + 1, hv.y * nrm);
        atomicAdd(ap + 2, hv.z * nrm);
        atomicAdd(ap + 3, hv.w * nrm);
    }
    __syncthreads();

    // ---- +bias +residual (x from LDS), LayerNorm(128), ReLU ----
    {
        const int c0 = lane << 2;  // 4 channels per lane, 128 per wave
        const float4 bi = *(const float4*)(bias   + c0);
        const float4 ga = *(const float4*)(gamma_ + c0);
        const float4 be = *(const float4*)(beta_  + c0);
        for (int r = wv; r < MROWS; r += 8) {
            const float4 ag = *(const float4*)&lAgg[r * HS + c0];
            const float4 xr = *(const float4*)&lXf[r * XSF + c0];
            float y0 = ag.x + bi.x + xr.x;
            float y1 = ag.y + bi.y + xr.y;
            float y2 = ag.z + bi.z + xr.z;
            float y3 = ag.w + bi.w + xr.w;
            float s = y0 + y1 + y2 + y3;
            #pragma unroll
            for (int off = 16; off > 0; off >>= 1) s += __shfl_xor(s, off, 32);
            float mu = s * (1.0f / 128.0f);
            float d0 = y0 - mu, d1 = y1 - mu, d2 = y2 - mu, d3 = y3 - mu;
            float v = d0 * d0 + d1 * d1 + d2 * d2 + d3 * d3;
            #pragma unroll
            for (int off = 16; off > 0; off >>= 1) v += __shfl_xor(v, off, 32);
            float rinv = rsqrtf(v * (1.0f / 128.0f) + LN_EPS);
            float4 o;
            o.x = fmaxf(0.f, d0 * rinv * ga.x + be.x);
            o.y = fmaxf(0.f, d1 * rinv * ga.y + be.y);
            o.z = fmaxf(0.f, d2 * rinv * ga.z + be.z);
            o.w = fmaxf(0.f, d3 * rinv * ga.w + be.w);
            *(float4*)(out + (long long)(row0 + r) * C_DIM + c0) = o;
        }
    }
}

extern "C" void kernel_launch(void* const* d_in, const int* in_sizes, int n_in,
                              void* d_out, int out_size, void* d_ws, size_t ws_size,
                              hipStream_t stream) {
    const float* x      = (const float*)d_in[0];
    const int*   ei     = (const int*)  d_in[1];   // edge_index [2, E] int32
    const float* Wm     = (const float*)d_in[2];
    const float* bias   = (const float*)d_in[3];
    const float* ew     = (const float*)d_in[4];
    const float* gamma_ = (const float*)d_in[5];
    const float* beta_  = (const float*)d_in[6];
    float* out = (float*)d_out;

    const int Etot    = in_sizes[1] / 2;           // 917504
    const int Ntot    = in_sizes[0] / C_DIM;       // 458752 nodes
    const int ngraphs = Ntot / NPG;                // 16384
    const int nblk    = ngraphs / GPB;             // 8192 workgroups

    hipLaunchKernelGGL(gcn_block_fused_wmma, dim3(nblk), dim3(256), 0, stream,
                       x, ei, Wm, bias, ew, gamma_, beta_, out, Etot);
}
